// LocalPredictor_8160437863087
// MI455X (gfx1250) — matrix operand order repaired and checked by reading, not compiled, not run on hardware
//
#include <hip/hip_runtime.h>
#include <hip/hip_bf16.h>

#define N_BITS 32
#define HIDDEN 128
#define HH 256
#define WW 256
#define BB 8

typedef __attribute__((ext_vector_type(16))) _Float16 v16h;
typedef __attribute__((ext_vector_type(8)))  _Float16 v8h;
typedef __attribute__((ext_vector_type(8)))  float    v8f;

// LDS layout (bytes):
//   zsf : [10][18][32] f32  = 23040   (raw async-copied z tile + halo)
//   w1f : 64 frags * 512 f16 = 65536  (p-chunk 0..7 x n-tile 0..7, per-lane B layout)
//   w2f : 8 frags * 512 f16  = 8192   (n2 0..1 x q-chunk 0..3)
//   hs  : 8 waves * 16*128 f16 = 32768
#define ZS_B    0
#define W1F_B   23040
#define W2F_B   (23040 + 65536)
#define HS_B    (23040 + 65536 + 8192)
#define SMEM_BYTES (23040 + 65536 + 8192 + 32768)

__global__ void __launch_bounds__(256)
local_predictor_wmma(const float* __restrict__ z,
                     const float* __restrict__ W1,
                     const float* __restrict__ b1,
                     const float* __restrict__ W2,
                     const float* __restrict__ b2,
                     float* __restrict__ out) {
    extern __shared__ __align__(16) char smemraw[];
    float*    zsf = (float*)   (smemraw + ZS_B);
    _Float16* w1f = (_Float16*)(smemraw + W1F_B);
    _Float16* w2f = (_Float16*)(smemraw + W2F_B);
    _Float16* hs  = (_Float16*)(smemraw + HS_B);

    const int tid = threadIdx.x;
    const int x0  = blockIdx.x * 16;   // 16 columns per block
    const int y0  = blockIdx.y * 8;    // 8 rows per block (1 per wave)
    const int b   = blockIdx.z;

    // global_prefetch_b8 on the first z row we will touch
    __builtin_prefetch(z + ((size_t)b * N_BITS) * (HH * WW) + (size_t)(y0 > 0 ? y0 - 1 : 0) * WW + x0, 0, 0);

    // ---- Stage z tile (with halo) into LDS as raw f32 via async global->LDS DMA ----
    // zsf layout: [pos(10x18)][c(32)] so a fragment's 8 channels are contiguous.
    for (int i = tid; i < 5760; i += 256) {
        int c   = i / 180;
        int pos = i - c * 180;          // 0..179 over 10x18 padded tile
        int yy  = pos / 18;
        int xx  = pos - yy * 18;
        int gy  = y0 + yy - 1;
        int gx  = x0 + xx - 1;
        unsigned lds_byte = (unsigned)(ZS_B + (pos * 32 + c) * 4);
        if (gy >= 0 && gy < HH && gx >= 0 && gx < WW) {
            int goff = (((b * N_BITS + c) * HH + gy) * WW + gx) * 4;   // byte offset (<2^31)
            asm volatile("global_load_async_to_lds_b32 %0, %1, %2"
                         :: "v"(lds_byte), "v"(goff), "s"(z)
                         : "memory");
        } else {
            zsf[pos * 32 + c] = 0.0f;   // zero-pad halo
        }
    }

    // ---- Stage W1 into per-lane B-fragment layout (f16) ----
    // B-layout (32x16 f16): lane half h holds K=8h+e (e<8), K=16+8h+(e-8) (e>=8); N = L%16
    // K == channel c at neighborhood position p9; W1 column = c*9 + p9 (k-outer, 9-inner)
    for (int i = tid; i < 32768; i += 256) {
        int f  = i >> 9;
        int L  = (i >> 4) & 31;
        int e  = i & 15;
        int pc = f >> 3;
        int n  = f & 7;
        int p9 = (pc < 4) ? pc : pc + 1;        // skip masked center position 4
        int c  = (e & 7) + ((e >> 3) << 4) + ((L >= 16) ? 8 : 0);
        int j  = L & 15;
        w1f[i] = (_Float16)W1[(n * 16 + j) * (9 * N_BITS) + c * 9 + p9];
    }

    // ---- Stage W2 into per-lane B-fragment layout (frag f2 = n2*4 + q) ----
    for (int i = tid; i < 4096; i += 256) {
        int f  = i >> 9;
        int L  = (i >> 4) & 31;
        int e  = i & 15;
        int n2 = f >> 2;
        int q  = f & 3;
        int d  = q * 32 + (e & 7) + ((e >> 3) << 4) + ((L >= 16) ? 8 : 0);
        int j  = L & 15;
        w2f[i] = (_Float16)W2[(n2 * 16 + j) * HIDDEN + d];
    }

    // Drain async tile DMA before anyone reads zsf, then block barrier.
    asm volatile("s_wait_asynccnt 0x0" ::: "memory");
    __syncthreads();

    const int lane = tid & 31;
    const int wave = tid >> 5;
    const int m    = lane & 15;       // A-row (pixel); also N column in D layout
    const int hiH  = lane >> 4;       // lane-group
    const int cA   = hiH ? 8 : 0;     // A-layout: lo half at K=cA.., hi half at K=cA+16..

    // ---- Build 8 A-fragments (16 pixels x 32 channels per neighborhood offset), f32->f16 ----
    v16h afrag[8];
#pragma unroll
    for (int pc = 0; pc < 8; ++pc) {
        int p9 = (pc < 4) ? pc : pc + 1;
        int dy = p9 / 3, dx = p9 - dy * 3;
        const float* zr = zsf + ((wave + dy) * 18 + (m + dx)) * 32;
        v16h a;
#pragma unroll
        for (int e = 0; e < 8; ++e) {
            a[e]     = (_Float16)zr[cA + e];
            a[e + 8] = (_Float16)zr[cA + 16 + e];
        }
        afrag[pc] = a;
    }

    // ---- GEMM1: H[16x128] = X * W1^T, zero-init acc, bias folded into ReLU epilogue ----
    _Float16* hsw = hs + wave * (16 * HIDDEN);
#pragma unroll
    for (int n = 0; n < 8; ++n) {
        v8f acc = {};                  // C = 0 (first WMMA can take inline 0)
#pragma unroll
        for (int pc = 0; pc < 8; ++pc) {
            v16h bfrag = *(const v16h*)(w1f + ((pc * 8 + n) << 9) + (lane << 4));
            acc = __builtin_amdgcn_wmma_f32_16x16x32_f16(
                false, afrag[pc], false, bfrag, (short)0, acc, false, false);
        }
        float bias = b1[n * 16 + m];   // D layout: this lane owns column N = n*16 + m
#pragma unroll
        for (int r = 0; r < 8; ++r) {
            float h = acc[r] + bias;
            h = h > 0.0f ? h : 0.0f;
            hsw[(r + hiH * 8) * HIDDEN + n * 16 + m] = (_Float16)h;   // [pixel][d]
        }
    }

    // ---- Re-load H as A-fragments (4 K-chunks of 32) ----
    v16h hfrag[4];
#pragma unroll
    for (int q = 0; q < 4; ++q) {
        const _Float16* hr = hsw + m * HIDDEN + q * 32;
        v8h lo = *(const v8h*)(hr + cA);
        v8h hi = *(const v8h*)(hr + cA + 16);
        hfrag[q] = __builtin_shufflevector(lo, hi, 0,1,2,3,4,5,6,7,8,9,10,11,12,13,14,15);
    }

    // ---- GEMM2: out[16x32] = H * W2^T + b2 ----
    const int gy = y0 + wave;
#pragma unroll
    for (int n2 = 0; n2 < 2; ++n2) {
        v8f acc = {};
#pragma unroll
        for (int q = 0; q < 4; ++q) {
            v16h bfrag = *(const v16h*)(w2f + ((n2 * 4 + q) << 9) + (lane << 4));
            acc = __builtin_amdgcn_wmma_f32_16x16x32_f16(
                false, hfrag[q], false, bfrag, (short)0, acc, false, false);
        }
        float bias = b2[n2 * 16 + m];
        int o = n2 * 16 + m;           // output channel (N in D layout)
#pragma unroll
        for (int r = 0; r < 8; ++r) {
            int mm = r + hiH * 8;      // pixel within strip (M in D layout)
            out[(((size_t)b * N_BITS + o) * HH + gy) * WW + (x0 + mm)] = acc[r] + bias;
        }
    }
}

extern "C" void kernel_launch(void* const* d_in, const int* in_sizes, int n_in,
                              void* d_out, int out_size, void* d_ws, size_t ws_size,
                              hipStream_t stream) {
    (void)in_sizes; (void)n_in; (void)d_ws; (void)ws_size; (void)out_size;
    const float* z  = (const float*)d_in[0];
    const float* W1 = (const float*)d_in[1];
    const float* b1 = (const float*)d_in[2];
    const float* W2 = (const float*)d_in[3];
    const float* b2 = (const float*)d_in[4];
    float* out = (float*)d_out;

    dim3 grid(WW / 16, HH / 8, BB);   // (16, 32, 8)
    dim3 block(256);
    local_predictor_wmma<<<grid, block, (size_t)SMEM_BYTES, stream>>>(z, W1, b1, W2, b2, out);
}
